// SimpleTransformerWithKVCache_86423331930112
// MI455X (gfx1250) — matrix operand order repaired and checked by reading, not compiled, not run on hardware
//
#include <hip/hip_runtime.h>
#include <hip/hip_bf16.h>
#include <math.h>

// ---------------- model constants (from the reference) ----------------
#define LAYERS 4
#define BATCH  16
#define HEADS  16
#define PAST   2048
#define QLEN   16
#define DHEAD  64
#define DMODEL 1024
#define DFF    4096
#define VOCAB  50257
#define SEQ    (PAST + QLEN)   // 2064 keys after append
#define SEQP   2080            // padded to a multiple of 32 for WMMA K-steps
#define ROWS   (BATCH * QLEN)  // 256 token rows

typedef __attribute__((ext_vector_type(16))) __bf16 v16bf;
typedef __attribute__((ext_vector_type(8)))  float  v8f;

// fp32 -> bf16 round-to-nearest-even (bit trick)
__device__ __forceinline__ unsigned short f2bf(float f) {
  union { float f; unsigned u; } c; c.f = f;
  unsigned u = c.u;
  return (unsigned short)((u + 0x7FFFu + ((u >> 16) & 1u)) >> 16);
}
// pack two fp32 -> one dword of 2x bf16
__device__ __forceinline__ unsigned pack2bf(float lo, float hi) {
  return (unsigned)f2bf(lo) | ((unsigned)f2bf(hi) << 16);
}

union ABFrag  { v16bf v; unsigned short h[16]; uint4 q[2]; };
union AccView { v8f   v; float          f[8];  };

// =====================================================================
// C[256,N] = act(A[256,K] @ W[K,N] + bias) (+residual), bf16 WMMA.
// K,N compile-time -> strided staging loads use *immediate* offsets.
// Block = 128 threads (4 wave32); tile 64x64; double-buffered LDS W tile
// in fragment-ready bf16 layout (per column+k-half: 16 contiguous values)
// so each lane's WMMA-B fragment is 2x ds_load_b128.
// Staging: thread = (column, 16-contiguous-K slab): 16 clause-able global
// loads -> pack in regs -> 2x ds_store_b128.
// KVMODE==1 scatters output into present-KV layout [B,H,SEQ,DH] @ PAST+qp.
// =====================================================================
template<int K, int N, int ACT, int KVMODE>
__global__ void __launch_bounds__(128)
gemm_bf16_wmma(const float* __restrict__ A, const float* __restrict__ W,
               const float* __restrict__ bias, const float* __restrict__ resid,
               float* __restrict__ C)
{
  __shared__ unsigned short wtile[2][64 * 2 * 16];   // 2 x 4 KB double buffer

  const int tid  = threadIdx.x;
  const int wave = tid >> 5;
  const int lane = tid & 31;
  const int half = lane >> 4;
  const int lr   = lane & 15;
  const int m0   = blockIdx.y * 64 + wave * 16;
  const int nb   = blockIdx.x * 64;

  // staging assignment: thread -> (column, k-half slab of 16 contiguous K)
  const int  cstage = tid & 63;
  const int  khalf  = tid >> 6;              // 0 or 1
  const int  colg   = nb + cstage;
  const bool cok    = (N % 64 == 0) ? true : (colg < N);
  const float* Wst  = W + colg;              // column base

  v8f acc[4] = {};
  const float* Arow = A + (size_t)(m0 + lr) * K;

  for (int kk = 0; kk < K; kk += 32) {
    const int buf = (kk >> 5) & 1;
    // ---- A fragment direct from global (b128 clauses)
    ABFrag a;
#pragma unroll
    for (int v = 0; v < 8; ++v) {
      int k0 = kk + half * 16 + (v << 1);
      a.h[2 * v]     = f2bf(Arow[k0]);
      a.h[2 * v + 1] = f2bf(Arow[k0 + 1]);
    }
    // ---- stage W[kk+khalf*16 .. +16) x column: 16 independent loads
    float wv[16];
#pragma unroll
    for (int j = 0; j < 16; ++j)
      wv[j] = cok ? Wst[(size_t)(kk + khalf * 16 + j) * N] : 0.f;
    if (cok && kk + 32 < K)
      __builtin_prefetch(Wst + (size_t)(kk + 32 + khalf * 16) * N, 0, 1);
    {
      unsigned short* dst = &wtile[buf][(cstage * 2 + khalf) * 16];
      uint4 p0, p1;
      p0.x = pack2bf(wv[0],  wv[1]);  p0.y = pack2bf(wv[2],  wv[3]);
      p0.z = pack2bf(wv[4],  wv[5]);  p0.w = pack2bf(wv[6],  wv[7]);
      p1.x = pack2bf(wv[8],  wv[9]);  p1.y = pack2bf(wv[10], wv[11]);
      p1.z = pack2bf(wv[12], wv[13]); p1.w = pack2bf(wv[14], wv[15]);
      ((uint4*)dst)[0] = p0;          // 2x ds_store_b128
      ((uint4*)dst)[1] = p1;
    }
    __syncthreads();                   // single barrier per K-step (dbl-buffered)
    // ---- 4 N-tiles per wave; B fragment = 2x ds_load_b128
#pragma unroll
    for (int t = 0; t < 4; ++t) {
      ABFrag bf;
      const uint4* bp = (const uint4*)&wtile[buf][((t * 16 + lr) * 2 + half) * 16];
      bf.q[0] = bp[0];
      bf.q[1] = bp[1];
      acc[t] = __builtin_amdgcn_wmma_f32_16x16x32_bf16(
          false, a.v, false, bf.v, (short)0, acc[t], false, false);
    }
  }

  // ---- epilogue: bias + residual + activation, scatter per mode
#pragma unroll
  for (int t = 0; t < 4; ++t) {
    int col = nb + t * 16 + lr;
    if ((N % 64 != 0) && col >= N) continue;
    AccView cv; cv.v = acc[t];
#pragma unroll
    for (int r = 0; r < 8; ++r) {
      int row = m0 + half * 8 + r;     // ISA C/D layout: VGPR r -> M = r + 8*half
      float val = cv.f[r];
      if (bias)  val += bias[col];
      if (resid) val += resid[(size_t)row * N + col];
      if (ACT == 1) val = 0.5f * val * (1.0f + erff(val * 0.70710678118654752f));
      if (KVMODE) {
        int b  = row >> 4, qp = row & 15;
        int hd = col >> 6, dh = col & 63;
        C[(((size_t)(b * HEADS + hd)) * SEQ + (PAST + qp)) * DHEAD + dh] = val;
      } else {
        C[(size_t)row * N + col] = val;
      }
    }
  }
}

// =====================================================================
// LayerNorm: one block (256 threads) per token row of 1024.
// =====================================================================
__global__ void __launch_bounds__(256)
layernorm_kernel(const float* __restrict__ X, const float* __restrict__ g,
                 const float* __restrict__ b, float* __restrict__ Y)
{
  __shared__ float red[256];
  const int row = blockIdx.x;
  const float* xr = X + (size_t)row * DMODEL;

  float s = 0.f;
  for (int i = threadIdx.x; i < DMODEL; i += 256) s += xr[i];
  red[threadIdx.x] = s; __syncthreads();
  for (int o = 128; o > 0; o >>= 1) {
    if (threadIdx.x < o) red[threadIdx.x] += red[threadIdx.x + o];
    __syncthreads();
  }
  float mean = red[0] * (1.0f / DMODEL);
  __syncthreads();

  float v = 0.f;
  for (int i = threadIdx.x; i < DMODEL; i += 256) { float d = xr[i] - mean; v += d * d; }
  red[threadIdx.x] = v; __syncthreads();
  for (int o = 128; o > 0; o >>= 1) {
    if (threadIdx.x < o) red[threadIdx.x] += red[threadIdx.x + o];
    __syncthreads();
  }
  float rstd = rsqrtf(red[0] * (1.0f / DMODEL) + 1e-5f);

  float* yr = Y + (size_t)row * DMODEL;
  for (int i = threadIdx.x; i < DMODEL; i += 256)
    yr[i] = (xr[i] - mean) * rstd * g[i] + b[i];
}

// =====================================================================
// x = emb[ids] + pos[pos_ids]
// =====================================================================
__global__ void embed_kernel(const int* __restrict__ ids, const int* __restrict__ pids,
                             const float* __restrict__ emb, const float* __restrict__ pos,
                             float* __restrict__ X)
{
  int i = blockIdx.x * blockDim.x + threadIdx.x;
  if (i >= ROWS * DMODEL) return;
  int row = i >> 10, d = i & (DMODEL - 1);
  X[i] = emb[(size_t)ids[row] * DMODEL + d] + pos[(size_t)pids[row] * DMODEL + d];
}

// =====================================================================
// Bulk copy past K/V [L,B,H,PAST,64] -> present caches [L,B,H,SEQ,64]
// (positions [0,PAST)). float4 vectorized, grid-stride; main BW consumer.
// =====================================================================
__global__ void __launch_bounds__(256)
kvcopy_kernel(const float* __restrict__ pk, const float* __restrict__ pv,
              float* __restrict__ outk, float* __restrict__ outv)
{
  const long long total = (long long)LAYERS * BATCH * HEADS * PAST * (DHEAD / 4);
  long long i = (long long)blockIdx.x * blockDim.x + threadIdx.x;
  const long long stride = (long long)gridDim.x * blockDim.x;
  for (; i < total; i += stride) {
    long long rowv = i >> 4;            // (l,b,h,p) row of 64 floats
    int j = (int)(i & 15);
    long long lbh = rowv >> 11;         // / PAST
    int p = (int)(rowv & (PAST - 1));
    const float4* sk = (const float4*)(pk + rowv * DHEAD) + j;
    const float4* sv = (const float4*)(pv + rowv * DHEAD) + j;
    float4* dk = (float4*)(outk + (lbh * SEQ + p) * DHEAD) + j;
    float4* dv = (float4*)(outv + (lbh * SEQ + p) * DHEAD) + j;
    *dk = *sk;
    *dv = *sv;
  }
}

// =====================================================================
// Fused attention: one workgroup per (b,h). 4 wave32.
// Phase 1: scores = (q @ K^T)/8 via WMMA -> dynamic LDS (full 16x2064
//          score matrix resident thanks to CDNA5's 320 KB/WGP LDS).
// Phase 2: row softmax in LDS (pad columns kept exactly 0).
// Phase 3: o = w @ V via WMMA; the 32x64 fp32 V tile per K-step is
//          staged with gfx1250 async DMA (global_load_async_to_lds_b128,
//          double-buffered; A-fragment built while DMA is in flight;
//          s_wait_asynccnt + one barrier per step). Tail keys zero-filled.
// =====================================================================
__global__ void __launch_bounds__(128)
attention_kernel(const float* __restrict__ Q, const float* __restrict__ Kc,
                 const float* __restrict__ Vc, float* __restrict__ O)
{
  extern __shared__ float smem[];
  float* sc  = smem;                    // [16][SEQP] scores
  float* red = smem + 16 * SEQP;        // 256 floats reduction scratch
  float* vt  = red + 256;               // 2 x [32][64] fp32 V tile (16 KB)

  const int bh = blockIdx.x;
  const int b = bh / HEADS, hd = bh % HEADS;
  const int wave = threadIdx.x >> 5, lane = threadIdx.x & 31;
  const int half = lane >> 4, lr = lane & 15;
  const float* Kbh = Kc + (size_t)bh * SEQ * DHEAD;
  const float* Vbh = Vc + (size_t)bh * SEQ * DHEAD;

  // ---- phase 1: scores ----
  ABFrag afr[2];            // q tile 16x64 as two 16x32 fragments (reused)
#pragma unroll
  for (int s = 0; s < 2; ++s) {
    const float* qr = Q + (size_t)(b * QLEN + lr) * DMODEL + hd * DHEAD + s * 32;
#pragma unroll
    for (int v = 0; v < 8; ++v) {
      int k0 = half * 16 + (v << 1);
      afr[s].h[2 * v]     = f2bf(qr[k0]);
      afr[s].h[2 * v + 1] = f2bf(qr[k0 + 1]);
    }
  }
  for (int tile = wave; tile < SEQ / 16; tile += 4) {   // 129 key tiles
    v8f acc = {};
    int key = tile * 16 + lr;
#pragma unroll
    for (int s = 0; s < 2; ++s) {
      ABFrag bf;            // B[k][n] = Kbh[key=n][k]: 16 contiguous floats/lane
      const float* kp = Kbh + (size_t)key * DHEAD + s * 32 + half * 16;
#pragma unroll
      for (int v = 0; v < 8; ++v) {
        bf.h[2 * v]     = f2bf(kp[2 * v]);
        bf.h[2 * v + 1] = f2bf(kp[2 * v + 1]);
      }
      acc = __builtin_amdgcn_wmma_f32_16x16x32_bf16(
          false, afr[s].v, false, bf.v, (short)0, acc, false, false);
    }
    AccView cv; cv.v = acc;
#pragma unroll
    for (int r = 0; r < 8; ++r)
      sc[(half * 8 + r) * SEQP + tile * 16 + lr] = cv.f[r] * 0.125f; // 1/sqrt(64)
  }
  for (int i = threadIdx.x; i < 16 * (SEQP - SEQ); i += 128) {       // zero pads
    int r = i / (SEQP - SEQ), c = i % (SEQP - SEQ);
    sc[r * SEQP + SEQ + c] = 0.f;
  }
  __syncthreads();

  // ---- phase 2: softmax (8 threads per row) ----
  {
    int row = threadIdx.x >> 3, sub = threadIdx.x & 7;
    float* srow = sc + row * SEQP;
    float mx = -3.4e38f;
    for (int kix = sub; kix < SEQ; kix += 8) mx = fmaxf(mx, srow[kix]);
    red[row * 8 + sub] = mx; __syncthreads();
    if (sub == 0) {
#pragma unroll
      for (int j = 1; j < 8; ++j) mx = fmaxf(mx, red[row * 8 + j]);
      red[row * 8] = mx;
    }
    __syncthreads();
    mx = red[row * 8];
    float sum = 0.f;
    for (int kix = sub; kix < SEQ; kix += 8) {
      float e = __expf(srow[kix] - mx);
      srow[kix] = e;
      sum += e;
    }
    red[128 + row * 8 + sub] = sum; __syncthreads();
    if (sub == 0) {
#pragma unroll
      for (int j = 1; j < 8; ++j) sum += red[128 + row * 8 + j];
      red[128 + row * 8] = sum;
    }
    __syncthreads();
    float inv = 1.0f / red[128 + row * 8];
    for (int kix = sub; kix < SEQ; kix += 8) srow[kix] *= inv;
  }
  __syncthreads();

  // ---- phase 3: o = w @ V (async-DMA staged, double-buffered V tile) ----
  {
    int n0 = wave * 16;               // each wave owns 16 of the 64 dh columns
    v8f acc = {};
    for (int kk = 0; kk < SEQP; kk += 32) {
      float* vbuf = vt + ((kk >> 5) & 1) * (32 * DHEAD);
      // 1) kick off async DMA of V[kk..kk+32) x [0..64) (16B chunks/lane)
      {
        int i = threadIdx.x;
#pragma unroll
        for (int c = 0; c < 4; ++c, i += 128) {
          int row = i >> 4, quad = i & 15;      // 32 rows x 16 quads of 16B
          int key = kk + row;
          float* dst = vbuf + row * DHEAD + quad * 4;
          if (key < SEQ) {
            unsigned ldsOff = (unsigned)(size_t)(void*)dst;   // LDS byte offset
            unsigned long long gaddr =
                (unsigned long long)(const void*)(Vbh + (size_t)key * DHEAD + quad * 4);
            asm volatile("global_load_async_to_lds_b128 %0, %1, off"
                         :: "v"(ldsOff), "v"(gaddr) : "memory");
          } else {
            float4 z = {0.f, 0.f, 0.f, 0.f};
            *(float4*)dst = z;        // zero-fill tail keys
          }
        }
      }
      // 2) build A fragment from LDS scores while the DMA is in flight
      ABFrag a;
#pragma unroll
      for (int v = 0; v < 8; ++v) {
        int k0 = kk + half * 16 + (v << 1);
        a.h[2 * v]     = f2bf(sc[lr * SEQP + k0]);
        a.h[2 * v + 1] = f2bf(sc[lr * SEQP + k0 + 1]);
      }
      // 3) wait DMA + barrier (double buffer -> one barrier per step)
      asm volatile("s_wait_asynccnt 0x0" ::: "memory");
      __syncthreads();
      // 4) B fragment from LDS V tile + WMMA
      ABFrag bf;
#pragma unroll
      for (int v = 0; v < 8; ++v) {
        int r0 = half * 16 + 2 * v;
        bf.h[2 * v]     = f2bf(vbuf[(r0)     * DHEAD + n0 + lr]);
        bf.h[2 * v + 1] = f2bf(vbuf[(r0 + 1) * DHEAD + n0 + lr]);
      }
      acc = __builtin_amdgcn_wmma_f32_16x16x32_bf16(
          false, a.v, false, bf.v, (short)0, acc, false, false);
    }
    AccView cv; cv.v = acc;
#pragma unroll
    for (int r = 0; r < 8; ++r)
      O[(size_t)(b * QLEN + half * 8 + r) * DMODEL + hd * DHEAD + n0 + lr] = cv.f[r];
  }
}

// =====================================================================
extern "C" void kernel_launch(void* const* d_in, const int* in_sizes, int n_in,
                              void* d_out, int out_size, void* d_ws, size_t ws_size,
                              hipStream_t stream)
{
  (void)in_sizes; (void)n_in; (void)out_size; (void)ws_size;
  const int*   ids  = (const int*)d_in[0];
  const int*   pids = (const int*)d_in[1];
  const float* pk   = (const float*)d_in[2];
  const float* pv   = (const float*)d_in[3];
  const float* emb  = (const float*)d_in[4];
  const float* pos  = (const float*)d_in[5];
  const float* wq   = (const float*)d_in[6];
  const float* wk   = (const float*)d_in[7];
  const float* wv   = (const float*)d_in[8];
  const float* wo   = (const float*)d_in[9];
  const float* bq   = (const float*)d_in[10];
  const float* bk   = (const float*)d_in[11];
  const float* bv   = (const float*)d_in[12];
  const float* bo   = (const float*)d_in[13];
  const float* ln1g = (const float*)d_in[14];
  const float* ln1b = (const float*)d_in[15];
  const float* ln2g = (const float*)d_in[16];
  const float* ln2b = (const float*)d_in[17];
  const float* w1   = (const float*)d_in[18];
  const float* b1   = (const float*)d_in[19];
  const float* w2   = (const float*)d_in[20];
  const float* b2   = (const float*)d_in[21];
  const float* lnfg = (const float*)d_in[22];
  const float* lnfb = (const float*)d_in[23];
  const float* lmw  = (const float*)d_in[24];
  const float* lmb  = (const float*)d_in[25];

  // d_out = logits [B,Q,V] ++ pres_k [L,B,H,SEQ,DH] ++ pres_v [L,B,H,SEQ,DH]
  float* logits = (float*)d_out;
  float* presk  = logits + (size_t)ROWS * VOCAB;
  float* presv  = presk + (size_t)LAYERS * BATCH * HEADS * SEQ * DHEAD;

  float* x  = (float*)d_ws;
  float* h  = x  + (size_t)ROWS * DMODEL;
  float* qb = h  + (size_t)ROWS * DMODEL;
  float* ob = qb + (size_t)ROWS * DMODEL;
  float* mb = ob + (size_t)ROWS * DMODEL;

  const size_t KVL = (size_t)BATCH * HEADS * SEQ * DHEAD;

  embed_kernel<<<dim3((ROWS * DMODEL + 255) / 256), dim3(256), 0, stream>>>(ids, pids, emb, pos, x);
  kvcopy_kernel<<<dim3(65536), dim3(256), 0, stream>>>(pk, pv, presk, presv);

  const dim3 blk(128);
  const dim3 g1((DMODEL + 63) / 64, ROWS / 64);
  const dim3 g2((DFF + 63) / 64, ROWS / 64);
  const size_t attn_smem = (size_t)(16 * SEQP + 256 + 2 * 32 * DHEAD) * sizeof(float);

  for (int l = 0; l < LAYERS; ++l) {
    const size_t WD = (size_t)l * DMODEL * DMODEL;
    layernorm_kernel<<<dim3(ROWS), dim3(256), 0, stream>>>(x, ln1g + l * DMODEL, ln1b + l * DMODEL, h);
    gemm_bf16_wmma<DMODEL, DMODEL, 0, 0><<<g1, blk, 0, stream>>>(h, wq + WD, bq + l * DMODEL, nullptr, qb);
    gemm_bf16_wmma<DMODEL, DMODEL, 0, 1><<<g1, blk, 0, stream>>>(h, wk + WD, bk + l * DMODEL, nullptr, presk + l * KVL);
    gemm_bf16_wmma<DMODEL, DMODEL, 0, 1><<<g1, blk, 0, stream>>>(h, wv + WD, bv + l * DMODEL, nullptr, presv + l * KVL);
    attention_kernel<<<dim3(BATCH * HEADS), blk, attn_smem, stream>>>(qb, presk + l * KVL, presv + l * KVL, ob);
    gemm_bf16_wmma<DMODEL, DMODEL, 0, 0><<<g1, blk, 0, stream>>>(ob, wo + WD, bo + l * DMODEL, x, x);
    layernorm_kernel<<<dim3(ROWS), dim3(256), 0, stream>>>(x, ln2g + l * DMODEL, ln2b + l * DMODEL, h);
    gemm_bf16_wmma<DMODEL, DFF, 1, 0><<<g2, blk, 0, stream>>>(h, w1 + (size_t)l * DMODEL * DFF, b1 + (size_t)l * DFF, nullptr, mb);
    gemm_bf16_wmma<DFF, DMODEL, 0, 0><<<g1, blk, 0, stream>>>(mb, w2 + (size_t)l * DFF * DMODEL, b2 + l * DMODEL, x, x);
  }

  layernorm_kernel<<<dim3(ROWS), dim3(256), 0, stream>>>(x, lnfg, lnfb, h);
  const dim3 gl((VOCAB + 63) / 64, ROWS / 64);
  gemm_bf16_wmma<DMODEL, VOCAB, 0, 0><<<gl, blk, 0, stream>>>(h, lmw, lmb, nullptr, logits);
}
